// FixedSparseMultiHead_53429393162983
// MI455X (gfx1250) — compile-verified, matrix operand order checked by reading, not run on hardware
//
#include <hip/hip_runtime.h>

typedef __attribute__((ext_vector_type(16))) _Float16 v16h;
typedef __attribute__((ext_vector_type(4)))  _Float16 v4h;
typedef __attribute__((ext_vector_type(8)))  float    v8f;

#define SEQ 2048
#define DM  768
#define NH  12
#define HD  64
#define WIN 102   // K_SP = int(2048*0.05) = 102

// ---------------------------------------------------------------------------
// fp32 -> fp16 conversion (vectorized x4)
// ---------------------------------------------------------------------------
__global__ void cvt_f32_f16(const float* __restrict__ src, _Float16* __restrict__ dst, int n4) {
  int i = blockIdx.x * blockDim.x + threadIdx.x;
  if (i < n4) {
    float4 v = ((const float4*)src)[i];
    v4h o;
    o[0] = (_Float16)v.x; o[1] = (_Float16)v.y;
    o[2] = (_Float16)v.z; o[3] = (_Float16)v.w;
    ((v4h*)dst)[i] = o;
  }
}

__device__ __forceinline__ v16h load_frag(const _Float16* p) {
  v16h f;
  ((float4*)&f)[0] = *(const float4*)(p);
  ((float4*)&f)[1] = *(const float4*)(p + 16);
  return f;
}

// ---------------------------------------------------------------------------
// WMMA GEMM:  out[m, n] = sum_k A[m,k] * W[n,k] + bias[n]
// A: [M, Din] f16 row-major, W: [768, Din] f16 row-major.
// Block = 128 threads (4 waves); block tile 128x64, each wave a 32x64 strip
// (2 A-fragments x 4 B-fragments -> 8 WMMA accumulators per K-step).
// K-loop is software-pipelined (double-buffered fragments).
// MODE 0: f16 out, layout [H][S][64]   (Q, K)
// MODE 1: f16 out, layout [H][64][S]   (V transposed)
// MODE 2: f32 out, layout [S][768]     (final projection -> d_out)
// ---------------------------------------------------------------------------
template<int MODE>
__global__ void gemm_wmma(const _Float16* __restrict__ A,
                          const _Float16* __restrict__ W,
                          const float* __restrict__ bias,
                          void* __restrict__ out,
                          int Din)
{
  const int lane = threadIdx.x & 31;
  const int wave = threadIdx.x >> 5;
  const int m0   = blockIdx.x * 128 + wave * 32;
  const int n0   = blockIdx.y * 64;
  const int hi   = lane >> 4;      // half-wave select (K sub-block)
  const int l16  = lane & 15;

  v8f zero = {0.f,0.f,0.f,0.f,0.f,0.f,0.f,0.f};
  v8f acc[8];                       // [row-group 0/1][nt 0..3]
  #pragma unroll
  for (int i = 0; i < 8; ++i) acc[i] = zero;

  const _Float16* arow0 = A + (size_t)(m0 + l16) * Din + hi * 8;
  const _Float16* arow1 = arow0 + (size_t)16 * Din;
  const _Float16* wbase = W + (size_t)(n0 + l16) * Din + hi * 8;

  // prologue: load first K-step fragments
  v16h a0 = load_frag(arow0);
  v16h a1 = load_frag(arow1);
  v16h b[4];
  #pragma unroll
  for (int nt = 0; nt < 4; ++nt) b[nt] = load_frag(wbase + (size_t)(nt * 16) * Din);

  for (int kk = 0; kk < Din; kk += 32) {
    // issue next-iteration loads before consuming current fragments
    int kn = (kk + 32 < Din) ? (kk + 32) : 0;   // last-iter loads are dead
    __builtin_prefetch(arow0 + kn + 32, 0, 1);
    v16h na0 = load_frag(arow0 + kn);
    v16h na1 = load_frag(arow1 + kn);
    v16h nb[4];
    #pragma unroll
    for (int nt = 0; nt < 4; ++nt)
      nb[nt] = load_frag(wbase + (size_t)(nt * 16) * Din + kn);

    #pragma unroll
    for (int nt = 0; nt < 4; ++nt) {
      acc[nt]     = __builtin_amdgcn_wmma_f32_16x16x32_f16(
          false, a0, false, b[nt], (short)0, acc[nt],     false, false);
      acc[4 + nt] = __builtin_amdgcn_wmma_f32_16x16x32_f16(
          false, a1, false, b[nt], (short)0, acc[4 + nt], false, false);
    }

    a0 = na0; a1 = na1;
    #pragma unroll
    for (int nt = 0; nt < 4; ++nt) b[nt] = nb[nt];
  }

  #pragma unroll
  for (int g = 0; g < 2; ++g) {
    #pragma unroll
    for (int nt = 0; nt < 4; ++nt) {
      int col = n0 + nt * 16 + l16;
      float bv = bias[col];
      #pragma unroll
      for (int v = 0; v < 8; ++v) {
        int row = m0 + g * 16 + v + hi * 8;
        float val = acc[g * 4 + nt][v] + bv;
        if (MODE == 0) {
          int h = col >> 6, d = col & 63;
          ((_Float16*)out)[(size_t)h * SEQ * HD + (size_t)row * HD + d] = (_Float16)val;
        } else if (MODE == 1) {
          int h = col >> 6, d = col & 63;
          ((_Float16*)out)[(size_t)(h * HD + d) * SEQ + row] = (_Float16)val;
        } else {
          ((float*)out)[(size_t)row * DM + col] = val;
        }
      }
    }
  }
}

// ---------------------------------------------------------------------------
// Sparse local attention. One wave per (16-query block, head).
// Keys for queries [q0, q0+15] with window 102 all lie in [q0-112, q0+15]
// (128 contiguous keys, 8 tiles). Invalid (key<0, key>q, key<q-101) -> -inf.
// ---------------------------------------------------------------------------
__global__ void attn_wmma(const _Float16* __restrict__ Qh, // [H][S][64]
                          const _Float16* __restrict__ Kh, // [H][S][64]
                          const _Float16* __restrict__ Vt, // [H][64][S]
                          _Float16* __restrict__ ctx)      // [S][768]
{
  __shared__ _Float16 plds[16 * 128];

  const int lane = threadIdx.x & 31;
  const int h    = blockIdx.y;
  const int q0   = blockIdx.x * 16;
  const int hi   = lane >> 4;
  const int l16  = lane & 15;
  const int kstart = q0 - 112;

  const _Float16* Qb = Qh + (size_t)h * SEQ * HD;
  const _Float16* Kb = Kh + (size_t)h * SEQ * HD;
  const _Float16* Vb = Vt + (size_t)h * HD * SEQ;

  // Q fragments for the two K-steps (HD=64 -> kk = 0, 32)
  v16h qa[2];
  {
    const _Float16* qrow = Qb + (size_t)(q0 + l16) * HD + hi * 8;
    #pragma unroll
    for (int i = 0; i < 2; ++i) {
      ((float4*)&qa[i])[0] = *(const float4*)(qrow + i * 32);
      ((float4*)&qa[i])[1] = *(const float4*)(qrow + i * 32 + 16);
    }
  }

  v8f zero = {0.f,0.f,0.f,0.f,0.f,0.f,0.f,0.f};
  v8f sacc[8];

  // scores = Q (16x64) x K^T (64 x 128) in 8 N-tiles
  #pragma unroll
  for (int t = 0; t < 8; ++t) {
    sacc[t] = zero;
    int key = kstart + t * 16 + l16;
    int kcl = key < 0 ? 0 : key;            // clamped load, masked later
    const _Float16* krow = Kb + (size_t)kcl * HD + hi * 8;
    #pragma unroll
    for (int i = 0; i < 2; ++i) {
      v16h kb;
      ((float4*)&kb)[0] = *(const float4*)(krow + i * 32);
      ((float4*)&kb)[1] = *(const float4*)(krow + i * 32 + 16);
      sacc[t] = __builtin_amdgcn_wmma_f32_16x16x32_f16(
          false, qa[i], false, kb, (short)0, sacc[t], false, false);
    }
  }

  // mask + scale; row max over 8 tiles and 16 lanes of the half-wave
  const float scale = 0.125f;               // 1/sqrt(64)
  float mx[8], sm[8];
  #pragma unroll
  for (int v = 0; v < 8; ++v) mx[v] = -3.0e38f;
  #pragma unroll
  for (int t = 0; t < 8; ++t) {
    int key = kstart + t * 16 + l16;
    #pragma unroll
    for (int v = 0; v < 8; ++v) {
      int q = q0 + v + hi * 8;
      bool valid = (key >= 0) && (key <= q) && (key >= q - (WIN - 1));
      float s = valid ? sacc[t][v] * scale : -3.0e38f;
      sacc[t][v] = s;
      mx[v] = fmaxf(mx[v], s);
    }
  }
  #pragma unroll
  for (int v = 0; v < 8; ++v) {
    #pragma unroll
    for (int off = 8; off >= 1; off >>= 1)
      mx[v] = fmaxf(mx[v], __shfl_xor(mx[v], off, 32));
  }
  #pragma unroll
  for (int v = 0; v < 8; ++v) sm[v] = 0.f;
  #pragma unroll
  for (int t = 0; t < 8; ++t) {
    #pragma unroll
    for (int v = 0; v < 8; ++v) {
      float p = __expf(sacc[t][v] - mx[v]);
      sacc[t][v] = p;
      sm[v] += p;
    }
  }
  #pragma unroll
  for (int v = 0; v < 8; ++v) {
    #pragma unroll
    for (int off = 8; off >= 1; off >>= 1)
      sm[v] += __shfl_xor(sm[v], off, 32);
  }

  // normalized probs -> LDS (transpose from C-layout to A-fragment layout)
  #pragma unroll
  for (int t = 0; t < 8; ++t) {
    #pragma unroll
    for (int v = 0; v < 8; ++v) {
      int row = v + hi * 8;
      plds[row * 128 + t * 16 + l16] = (_Float16)(sacc[t][v] / sm[v]);
    }
  }
  __syncthreads();

  // ctx = P (16x128) x Vg (128x64): 4 K-steps x 4 N-tiles
  v8f cacc[4];
  #pragma unroll
  for (int i = 0; i < 4; ++i) cacc[i] = zero;

  #pragma unroll
  for (int ks = 0; ks < 4; ++ks) {
    v16h pa;
    const _Float16* pr = plds + l16 * 128 + ks * 32 + hi * 8;
    ((float4*)&pa)[0] = *(const float4*)(pr);
    ((float4*)&pa)[1] = *(const float4*)(pr + 16);
    #pragma unroll
    for (int nt = 0; nt < 4; ++nt) {
      int n = nt * 16 + l16;
      int k0 = kstart + ks * 32 + hi * 8;   // 8-aligned: group never straddles 0
      int k1 = k0 + 16;
      const _Float16* p0 = Vb + (size_t)n * SEQ + (k0 < 0 ? 0 : k0);
      const _Float16* p1 = Vb + (size_t)n * SEQ + (k1 < 0 ? 0 : k1);
      v16h vb;
      ((float4*)&vb)[0] = *(const float4*)p0;
      ((float4*)&vb)[1] = *(const float4*)p1;
      cacc[nt] = __builtin_amdgcn_wmma_f32_16x16x32_f16(
          false, pa, false, vb, (short)0, cacc[nt], false, false);
    }
  }

  // store ctx row-major [S][768]
  #pragma unroll
  for (int nt = 0; nt < 4; ++nt) {
    int d = nt * 16 + l16;
    #pragma unroll
    for (int v = 0; v < 8; ++v) {
      int row = q0 + v + hi * 8;
      ctx[(size_t)row * DM + h * HD + d] = (_Float16)cacc[nt][v];
    }
  }
}

// ---------------------------------------------------------------------------
extern "C" void kernel_launch(void* const* d_in, const int* in_sizes, int n_in,
                              void* d_out, int out_size, void* d_ws, size_t ws_size,
                              hipStream_t stream) {
  const float* x  = (const float*)d_in[0];
  const float* Wq = (const float*)d_in[1];
  const float* bq = (const float*)d_in[2];
  const float* Wk = (const float*)d_in[3];
  const float* bk = (const float*)d_in[4];
  const float* Wv = (const float*)d_in[5];
  const float* bv = (const float*)d_in[6];
  const float* Wo = (const float*)d_in[7];
  const float* bo = (const float*)d_in[8];
  // d_in[9] (indices), d_in[10] (index_mask): fixed causal local pattern, hardcoded.

  _Float16* ws  = (_Float16*)d_ws;
  _Float16* xh  = ws;                       // [S][768]
  _Float16* wqh = xh  + (size_t)SEQ * DM;   // [768][768]
  _Float16* wkh = wqh + (size_t)DM * DM;
  _Float16* wvh = wkh + (size_t)DM * DM;
  _Float16* woh = wvh + (size_t)DM * DM;
  _Float16* Qh  = woh + (size_t)DM * DM;    // [H][S][64]
  _Float16* Kh  = Qh  + (size_t)NH * SEQ * HD;
  _Float16* Vt  = Kh  + (size_t)NH * SEQ * HD;  // [H][64][S]
  _Float16* ctx = Vt  + (size_t)NH * SEQ * HD;  // [S][768]

  // fp32 -> fp16
  {
    int n4 = (SEQ * DM) / 4;
    cvt_f32_f16<<<(n4 + 255) / 256, 256, 0, stream>>>(x, xh, n4);
    int w4 = (DM * DM) / 4;
    cvt_f32_f16<<<(w4 + 255) / 256, 256, 0, stream>>>(Wq, wqh, w4);
    cvt_f32_f16<<<(w4 + 255) / 256, 256, 0, stream>>>(Wk, wkh, w4);
    cvt_f32_f16<<<(w4 + 255) / 256, 256, 0, stream>>>(Wv, wvh, w4);
    cvt_f32_f16<<<(w4 + 255) / 256, 256, 0, stream>>>(Wo, woh, w4);
  }

  dim3 ggrid(SEQ / 128, DM / 64);
  gemm_wmma<0><<<ggrid, 128, 0, stream>>>(xh, wqh, bq, (void*)Qh, DM);
  gemm_wmma<0><<<ggrid, 128, 0, stream>>>(xh, wkh, bk, (void*)Kh, DM);
  gemm_wmma<1><<<ggrid, 128, 0, stream>>>(xh, wvh, bv, (void*)Vt, DM);

  attn_wmma<<<dim3(SEQ / 16, NH), 32, 0, stream>>>(Qh, Kh, Vt, ctx);

  gemm_wmma<2><<<ggrid, 128, 0, stream>>>(ctx, woh, bo, d_out, DM);
}
